// FalseDependencyBlock_20572893348682
// MI455X (gfx1250) — matrix operand order repaired and checked by reading, not compiled, not run on hardware
//
#include <hip/hip_runtime.h>

// y = x @ W.T  — x:(32768,1024) f32, W:(1024,1024) f32, y:(32768,1024) f32
// bf16x3 split GEMM using v_wmma_f32_16x16x32_bf16 (fp32 accumulate).
// Software-pipelined; each wave owns 32(M) x 64(N) -> 24 WMMA per K-step,
// B fragments reused across 2 M-tiles (ds_load per WMMA ratio 1.0).

#define CH     1024
#define NROWS  32768
#define BM     256
#define BN     64
#define BK     32
#define LDA    (BK + 8)      // padded LDS row stride (bf16 elems): 80 B, 16B-aligned
#define NT     (BN / 16)     // 4 N-tiles of 16 per wave
#define MT     2             // 2 M-tiles of 16 per wave

typedef __bf16 bf16_t;
typedef __attribute__((ext_vector_type(16))) __bf16 v16bf;
typedef __attribute__((ext_vector_type(8)))  __bf16 v8bf;
typedef __attribute__((ext_vector_type(8)))  float  v8f;

__device__ __forceinline__ v16bf concat8(v8bf a, v8bf b) {
    return __builtin_shufflevector(a, b, 0,1,2,3,4,5,6,7,8,9,10,11,12,13,14,15);
}

__global__ __launch_bounds__(256)
void FalseDependencyBlock_gemm_bf16x3(const float* __restrict__ X,
                                      const float* __restrict__ W,
                                      float* __restrict__ Y)
{
    __shared__ bf16_t Ahi[BM][LDA];
    __shared__ bf16_t Alo[BM][LDA];
    __shared__ bf16_t Bhi[BN][LDA];
    __shared__ bf16_t Blo[BN][LDA];

    const int tid  = threadIdx.x;
    const int wave = tid >> 5;          // 0..7, owns rows [32*wave, 32*wave+32)
    const int lane = tid & 31;
    const int sel  = lane >> 4;         // half-wave select (0/1)
    const int l16  = lane & 15;

    const int blockM = blockIdx.x * BM;
    const int blockN = blockIdx.y * BN;

    // Per-thread staging addresses (fixed across iterations).
    // A tile: 256 rows x 32 cols = 2048 float4, 8 per thread.
    // B tile:  64 rows x 32 cols =  512 float4, 2 per thread.
    int arow[8], acol[8], brow[2], bcol[2];
#pragma unroll
    for (int i = 0; i < 8; ++i) {
        int fid = tid + 256 * i;
        arow[i] = fid >> 3;             // 8 float4 per 32-float row
        acol[i] = (fid & 7) << 2;
    }
#pragma unroll
    for (int i = 0; i < 2; ++i) {
        int fid = tid + 256 * i;
        brow[i] = fid >> 3;
        bcol[i] = (fid & 7) << 2;
    }

    const v8f vzero = {0.f,0.f,0.f,0.f,0.f,0.f,0.f,0.f};
    v8f acc[MT][NT];
#pragma unroll
    for (int m = 0; m < MT; ++m)
#pragma unroll
        for (int t = 0; t < NT; ++t) acc[m][t] = vzero;

    auto load_tiles = [&](int k0, float4* ra, float4* rb) {
#pragma unroll
        for (int i = 0; i < 8; ++i)
            ra[i] = *reinterpret_cast<const float4*>(
                X + (size_t)(blockM + arow[i]) * CH + k0 + acol[i]);
#pragma unroll
        for (int i = 0; i < 2; ++i)
            rb[i] = *reinterpret_cast<const float4*>(
                W + (size_t)(blockN + brow[i]) * CH + k0 + bcol[i]);
    };

    // f32 -> (bf16 hi, bf16 lo) with RNE casts
    auto split1 = [](float f, bf16_t* hp, bf16_t* lp) {
        bf16_t hi = (bf16_t)f;
        *hp = hi;
        *lp = (bf16_t)(f - (float)hi);
    };
    auto stage_tiles = [&](const float4* ra, const float4* rb) {
#pragma unroll
        for (int i = 0; i < 8; ++i) {
            int r = arow[i], c = acol[i];
            split1(ra[i].x, &Ahi[r][c+0], &Alo[r][c+0]);
            split1(ra[i].y, &Ahi[r][c+1], &Alo[r][c+1]);
            split1(ra[i].z, &Ahi[r][c+2], &Alo[r][c+2]);
            split1(ra[i].w, &Ahi[r][c+3], &Alo[r][c+3]);
        }
#pragma unroll
        for (int i = 0; i < 2; ++i) {
            int r = brow[i], c = bcol[i];
            split1(rb[i].x, &Bhi[r][c+0], &Blo[r][c+0]);
            split1(rb[i].y, &Bhi[r][c+1], &Blo[r][c+1]);
            split1(rb[i].z, &Bhi[r][c+2], &Blo[r][c+2]);
            split1(rb[i].w, &Bhi[r][c+3], &Blo[r][c+3]);
        }
    };

    // ---- prologue: stage tile 0 ----
    float4 regA[8], regB[2];
    load_tiles(0, regA, regB);
    stage_tiles(regA, regB);
    __syncthreads();

#pragma unroll 1
    for (int k0 = 0; k0 < CH; k0 += BK) {
        const bool more = (k0 + BK) < CH;

        // issue global loads for the NEXT tile before computing this one
        if (more) {
            load_tiles(k0 + BK, regA, regB);
            if (k0 + 2 * BK < CH) {
                __builtin_prefetch(X + (size_t)(blockM + arow[0]) * CH
                                     + k0 + 2 * BK + acol[0], 0, 0);
                __builtin_prefetch(W + (size_t)(blockN + brow[0]) * CH
                                     + k0 + 2 * BK + bcol[0], 0, 0);
            }
        }

        // ---- A fragments (16x32 bf16): lane l: row = l&15,
        //      halves 0..7 -> K = 8*sel + 0..7, halves 8..15 -> K = 16+8*sel + 0..7
        v16bf ahi[MT], alo[MT];
#pragma unroll
        for (int m = 0; m < MT; ++m) {
            const int am = wave * 32 + m * 16 + l16;
            ahi[m] = concat8(
                *reinterpret_cast<const v8bf*>(&Ahi[am][8 * sel]),
                *reinterpret_cast<const v8bf*>(&Ahi[am][16 + 8 * sel]));
            alo[m] = concat8(
                *reinterpret_cast<const v8bf*>(&Alo[am][8 * sel]),
                *reinterpret_cast<const v8bf*>(&Alo[am][16 + 8 * sel]));
        }

        // ---- B fragments (32x16 bf16): lane l: col = l&15,
        //      halves 0..7 -> K = 8*sel + 0..7, halves 8..15 -> K = 16+8*sel + 0..7
#pragma unroll
        for (int t = 0; t < NT; ++t) {
            const int bn = t * 16 + l16;
            const int bk = 8 * sel;
            v16bf bhi = concat8(
                *reinterpret_cast<const v8bf*>(&Bhi[bn][bk]),
                *reinterpret_cast<const v8bf*>(&Bhi[bn][bk + 16]));
            v16bf blo = concat8(
                *reinterpret_cast<const v8bf*>(&Blo[bn][bk]),
                *reinterpret_cast<const v8bf*>(&Blo[bn][bk + 16]));

            // acc += Ahi*Bhi + Ahi*Blo + Alo*Bhi   (bf16x3 fp32 emulation)
#pragma unroll
            for (int m = 0; m < MT; ++m) {
                acc[m][t] = __builtin_amdgcn_wmma_f32_16x16x32_bf16(
                    false, ahi[m], false, bhi, (short)0, acc[m][t], false, false);
                acc[m][t] = __builtin_amdgcn_wmma_f32_16x16x32_bf16(
                    false, ahi[m], false, blo, (short)0, acc[m][t], false, false);
                acc[m][t] = __builtin_amdgcn_wmma_f32_16x16x32_bf16(
                    false, alo[m], false, bhi, (short)0, acc[m][t], false, false);
            }
        }

        __syncthreads();                 // everyone done reading LDS tile k

        if (more) {
            stage_tiles(regA, regB);     // split + ds_store tile k+1
            __syncthreads();
        }
    }

    // ---- epilogue: C/D layout: VGPR r, lanes 0-15 -> M=r, lanes 16-31 -> M=r+8
#pragma unroll
    for (int m = 0; m < MT; ++m) {
#pragma unroll
        for (int t = 0; t < NT; ++t) {
            const int col = blockN + t * 16 + l16;
#pragma unroll
            for (int r = 0; r < 8; ++r) {
                const int row = blockM + wave * 32 + m * 16 + 8 * sel + r;
                Y[(size_t)row * CH + col] = acc[m][t][r];
            }
        }
    }
}

extern "C" void kernel_launch(void* const* d_in, const int* in_sizes, int n_in,
                              void* d_out, int out_size, void* d_ws, size_t ws_size,
                              hipStream_t stream) {
    const float* X = (const float*)d_in[0];   // (32768, 1024) f32
    const float* W = (const float*)d_in[1];   // (1024, 1024) f32
    float* Y = (float*)d_out;                 // (32768, 1024) f32

    dim3 grid(NROWS / BM, CH / BN);           // 128 x 16 workgroups
    dim3 block(256);                          // 8 wave32
    FalseDependencyBlock_gemm_bf16x3<<<grid, block, 0, stream>>>(X, W, Y);
}
